// DotProductAttention_50165218017960
// MI455X (gfx1250) — matrix operand order repaired and checked by reading, not compiled, not run on hardware
//
#include <hip/hip_runtime.h>
#include <hip/hip_bf16.h>

typedef __attribute__((ext_vector_type(16))) __bf16 v16bf;
typedef __attribute__((ext_vector_type(8)))  __bf16 v8bf;
typedef __attribute__((ext_vector_type(4)))  __bf16 v4bf;
typedef __attribute__((ext_vector_type(8)))  float  v8f;

#define B_    8
#define LQ    4096
#define LK    4096
#define DH    64
#define BK    64          // keys per chunk
#define WAVES 4
#define BQW   16          // q rows per wave
#define MASKV (-1.0e6f)

// LDS row strides in bf16 elements: 64 + 8 pad = 72 halfs = 144 B = 36 banks.
// key*36 mod 64 is distinct for 16 consecutive rows and =0 mod 4 -> the 16B
// vector reads of the 16 lanes hit disjoint 4-bank spans. 144 % 16 == 0 keeps
// 16B alignment for b128 loads.
#define KSTR  72
#define VSTR  72
#define PSTR  72

union U16 { v16bf v; v8bf h[2]; };

__device__ inline v16bf ld16_contig(const __bf16* p) {
  U16 u;
  u.h[0] = *(const v8bf*)(p);
  u.h[1] = *(const v8bf*)(p + 8);
  return u.v;
}

__device__ inline v8f wmma_bf16(v16bf a, v16bf b, v8f c) {
  return __builtin_amdgcn_wmma_f32_16x16x32_bf16(
      false, a, false, b, (short)0, c, false, false);
}

// ---------------- prep kernels: f32 -> bf16, V transposed ----------------

__global__ __launch_bounds__(256) void prep_k_bf16(
    const float* __restrict__ src, __bf16* __restrict__ dst, int n4) {
  int i = blockIdx.x * blockDim.x + threadIdx.x;
  if (i < n4) {
    float4 f = ((const float4*)src)[i];
    v4bf o;
    o[0] = (__bf16)f.x; o[1] = (__bf16)f.y; o[2] = (__bf16)f.z; o[3] = (__bf16)f.w;
    ((v4bf*)dst)[i] = o;
  }
}

__global__ __launch_bounds__(256) void prep_v_transpose(
    const float* __restrict__ V, __bf16* __restrict__ Vt) {
  // one block per (b, 64-key tile); tile = 64 keys x 64 dims
  __shared__ __bf16 t[DH][72];    // [d][key], padded
  const int b  = blockIdx.x / (LK / 64);
  const int k0 = (blockIdx.x % (LK / 64)) * 64;
  const int tid = threadIdx.x;
#pragma unroll
  for (int r = 0; r < 4; ++r) {
    const int idx = tid + r * 256;          // 0..1023 = 64 keys x 16 float4
    const int k   = idx >> 4;
    const int d4  = (idx & 15) * 4;
    const float4 f = *(const float4*)(V + ((size_t)b * LK + k0 + k) * DH + d4);
    t[d4 + 0][k] = (__bf16)f.x;
    t[d4 + 1][k] = (__bf16)f.y;
    t[d4 + 2][k] = (__bf16)f.z;
    t[d4 + 3][k] = (__bf16)f.w;
  }
  __syncthreads();
#pragma unroll
  for (int r = 0; r < 2; ++r) {
    const int idx = tid + r * 256;          // 0..511 = 64 d-rows x 8 segs
    const int d   = idx >> 3;
    const int k8  = (idx & 7) * 8;
    *(v8bf*)(Vt + ((size_t)b * DH + d) * LK + k0 + k8) = *(const v8bf*)&t[d][k8];
  }
}

// ---------------- flash-attention forward ----------------

__global__ __launch_bounds__(128) void fa_fwd_kernel(
    const __bf16* __restrict__ Kb,   // [B][LK][DH] bf16 row-major
    const __bf16* __restrict__ Vt,   // [B][DH][LK] bf16 (transposed)
    const float*  __restrict__ Qg,
    const long long* __restrict__ vlen,
    float* __restrict__ Og)
{
  __shared__ __bf16 lK[2][BK * KSTR];         // [key][d], double buffered
  __shared__ __bf16 lV[2][DH * VSTR];         // [d][key], double buffered
  __shared__ __bf16 lP[WAVES][BQW * PSTR];    // per-wave P scratch [m][k]

  const int tid  = threadIdx.x;
  const int wave = tid >> 5;
  const int lane = tid & 31;
  const int col  = lane & 15;
  const int hseg = lane >> 4;
  const int kb   = hseg * 16;     // per-lane K base inside A-operand halves

  const int b  = blockIdx.x / (LQ / (BQW * WAVES));
  const int qt = blockIdx.x % (LQ / (BQW * WAVES));
  const int q0 = qt * (BQW * WAVES) + wave * BQW;

  const int   vl    = (int)vlen[b];
  const float scale = 0.125f;     // 1/sqrt(64)

  const unsigned long long kbase = (unsigned long long)(uintptr_t)(Kb + (size_t)b * LK * DH);
  const unsigned long long vbase = (unsigned long long)(uintptr_t)(Vt + (size_t)b * DH * LK);

  // async stage of one 64-key chunk into buffer `buf`
  auto stage = [&](int kc, int buf) {
#pragma unroll
    for (int r = 0; r < 4; ++r) {
      const int seg = tid + r * 128;     // 0..511 = 64 rows x 8 x16B segs
      const int row = seg >> 3;          // key (for K) / d (for Vt)
      const int s8  = (seg & 7) * 8;     // halfs within row
      const unsigned ldsK = (unsigned)(uintptr_t)&lK[buf][row * KSTR + s8];
      const unsigned offK = (unsigned)(((kc + row) * DH + s8) * 2);
      asm volatile("global_load_async_to_lds_b128 %0, %1, %2"
                   :: "v"(ldsK), "v"(offK), "s"(kbase) : "memory");
      const unsigned ldsV = (unsigned)(uintptr_t)&lV[buf][row * VSTR + s8];
      const unsigned offV = (unsigned)((row * LK + kc + s8) * 2);
      asm volatile("global_load_async_to_lds_b128 %0, %1, %2"
                   :: "v"(ldsV), "v"(offV), "s"(vbase) : "memory");
    }
  };

  // ---- preload Q tile as two bf16 A-operands (16x32 each) ----
  const float* qrow = Qg + ((size_t)b * LQ + q0 + col) * DH;
  v16bf a0, a1;
#pragma unroll
  for (int i = 0; i < 8; ++i) {
    a0[i]     = (__bf16)qrow[kb + i];
    a0[i + 8] = (__bf16)qrow[kb + 8 + i];
    a1[i]     = (__bf16)qrow[32 + kb + i];
    a1[i + 8] = (__bf16)qrow[32 + kb + 8 + i];
  }

  v8f o[4] = {};
  float rmax[8], rsum[8];
#pragma unroll
  for (int j = 0; j < 8; ++j) { rmax[j] = -3.0e38f; rsum[j] = 0.0f; }

  stage(0, 0);

  int buf = 0;
  for (int kc = 0; kc < LK; kc += BK, buf ^= 1) {
    // own async copies done -> barrier makes every wave's copies visible
    asm volatile("s_wait_asynccnt 0x0" ::: "memory");
    __syncthreads();
    if (kc + BK < LK) stage(kc + BK, buf ^ 1);

    // ---- S = Q K^T : four 16x16 key tiles ----
    v8f s[4] = {};
    const __bf16* kbuf = &lK[buf][0];
#pragma unroll
    for (int t4 = 0; t4 < 4; ++t4) {
      const __bf16* r = &kbuf[(t4 * 16 + col) * KSTR + hseg * 16];
      s[t4] = wmma_bf16(a0, ld16_contig(r), s[t4]);
      s[t4] = wmma_bf16(a1, ld16_contig(r + 32), s[t4]);
    }

    // ---- scale + length-mask + online softmax ----
    const int keyc = kc + col;
    const bool m0 = (keyc      ) >= vl;
    const bool m1 = (keyc + 16 ) >= vl;
    const bool m2 = (keyc + 32 ) >= vl;
    const bool m3 = (keyc + 48 ) >= vl;
    __bf16* pw = &lP[wave][0];
#pragma unroll
    for (int j = 0; j < 8; ++j) {
      float x0 = m0 ? MASKV : s[0][j] * scale;
      float x1 = m1 ? MASKV : s[1][j] * scale;
      float x2 = m2 ? MASKV : s[2][j] * scale;
      float x3 = m3 ? MASKV : s[3][j] * scale;
      float mx = fmaxf(fmaxf(x0, x1), fmaxf(x2, x3));
#pragma unroll
      for (int off = 1; off < 16; off <<= 1)
        mx = fmaxf(mx, __shfl_xor(mx, off, 32));
      const float nm   = fmaxf(rmax[j], mx);
      const float corr = __expf(rmax[j] - nm);
      rmax[j] = nm;
      const float p0 = __expf(x0 - nm);
      const float p1 = __expf(x1 - nm);
      const float p2 = __expf(x2 - nm);
      const float p3 = __expf(x3 - nm);
      rsum[j] = rsum[j] * corr + ((p0 + p1) + (p2 + p3));
      o[0][j] *= corr; o[1][j] *= corr; o[2][j] *= corr; o[3][j] *= corr;
      const int row = (j + 8 * hseg) * PSTR;
      pw[row + col]      = (__bf16)p0;
      pw[row + 16 + col] = (__bf16)p1;
      pw[row + 32 + col] = (__bf16)p2;
      pw[row + 48 + col] = (__bf16)p3;
    }
    // in-wave LDS RAW: DS ops are in-order per wave; wait + compiler barrier
    asm volatile("s_wait_dscnt 0x0" ::: "memory");

    // ---- gather P as two 16x32 A-operands (keys 0..31 and 32..63) ----
    v16bf pa0, pa1;
    {
      const __bf16* pr = &pw[col * PSTR];
      U16 u0, u1;
      u0.h[0] = *(const v8bf*)(pr + kb);
      u0.h[1] = *(const v8bf*)(pr + kb + 8);
      u1.h[0] = *(const v8bf*)(pr + 32 + kb);
      u1.h[1] = *(const v8bf*)(pr + 32 + kb + 8);
      pa0 = u0.v; pa1 = u1.v;
    }

    // ---- O += P V : 4 output column blocks, two K=32 halves ----
    const __bf16* vbuf = &lV[buf][0];
#pragma unroll
    for (int nb = 0; nb < 4; ++nb) {
      const __bf16* vr = &vbuf[(nb * 16 + col) * VSTR + hseg * 16];
      o[nb] = wmma_bf16(pa0, ld16_contig(vr), o[nb]);
      o[nb] = wmma_bf16(pa1, ld16_contig(vr + 32), o[nb]);
    }

    __syncthreads();
  }

  // ---- finalize: reduce row sums, normalize, store ----
  float* obase = Og + ((size_t)b * LQ + q0) * DH;
#pragma unroll
  for (int j = 0; j < 8; ++j) {
    float ssum = rsum[j];
#pragma unroll
    for (int off = 1; off < 16; off <<= 1)
      ssum += __shfl_xor(ssum, off, 32);
    const float inv = 1.0f / ssum;
    float* orow = obase + (size_t)(j + 8 * hseg) * DH;
    orow[col]      = o[0][j] * inv;
    orow[16 + col] = o[1][j] * inv;
    orow[32 + col] = o[2][j] * inv;
    orow[48 + col] = o[3][j] * inv;
  }
}

extern "C" void kernel_launch(void* const* d_in, const int* in_sizes, int n_in,
                              void* d_out, int out_size, void* d_ws, size_t ws_size,
                              hipStream_t stream) {
  const float*     Q  = (const float*)d_in[0];
  const float*     K  = (const float*)d_in[1];
  const float*     V  = (const float*)d_in[2];
  const long long* vl = (const long long*)d_in[3];  // reference dtype int64
  float* O = (float*)d_out;

  __bf16* Kb = (__bf16*)d_ws;                                   // 4 MB
  __bf16* Vt = (__bf16*)((char*)d_ws + (size_t)B_ * LK * DH * 2); // 4 MB

  const int n4 = B_ * LK * DH / 4;
  prep_k_bf16<<<dim3((n4 + 255) / 256), dim3(256), 0, stream>>>(K, Kb, n4);
  prep_v_transpose<<<dim3(B_ * (LK / 64)), dim3(256), 0, stream>>>(V, Vt);

  dim3 grid(B_ * (LQ / (BQW * WAVES)));   // 512 workgroups
  dim3 block(128);                        // 4 waves
  fa_fwd_kernel<<<grid, block, 0, stream>>>(Kb, Vt, Q, vl, O);
}
